// PairToSequenceTE_32031866094097
// MI455X (gfx1250) — matrix operand order repaired and checked by reading, not compiled, not run on hardware
//
#include <hip/hip_runtime.h>
#include <hip/hip_bf16.h>

typedef unsigned short u16;
typedef unsigned int   u32;

typedef __bf16 bf16x16 __attribute__((ext_vector_type(16)));
typedef float  f32x8   __attribute__((ext_vector_type(8)));
typedef unsigned int u32x4 __attribute__((ext_vector_type(4)));
typedef int          i32x8 __attribute__((ext_vector_type(8)));
typedef int          i32x4 __attribute__((ext_vector_type(4)));

union ABu { uint4 q[2]; bf16x16 v; };

#if __has_builtin(__builtin_amdgcn_wave_barrier)
#define WAVE_BARRIER() __builtin_amdgcn_wave_barrier()
#else
#define WAVE_BARRIER() asm volatile("" ::: "memory")
#endif

static constexpr int L_DIM = 1024;
static constexpr int CZ    = 128;
static constexpr int CS    = 1024;
static constexpr int KCAT  = 2 * CZ + CS;   // 1280

// Software fallback (round-to-nearest-even) for host-independent paths.
__device__ __forceinline__ u16 f2bf_sw(float f) {
  u32 u = __builtin_bit_cast(u32, f);
  u32 r = (u + 0x7FFFu + ((u >> 16) & 1u)) >> 16;
  return (u16)r;
}

// Packed f32x2 -> bf16x2: builtin if declared, else the gfx1250 HW packed
// converter via inline asm (1 VALU op instead of ~6).
#if __has_builtin(__builtin_amdgcn_cvt_pk_bf16_f32)
typedef __bf16 bf16x2 __attribute__((ext_vector_type(2)));
__device__ __forceinline__ u32 pack_bf16(float a, float b) {
  bf16x2 p = __builtin_amdgcn_cvt_pk_bf16_f32(a, b);
  return __builtin_bit_cast(u32, p);
}
#else
__device__ __forceinline__ u32 pack_bf16(float a, float b) {
  u32 r;
  asm("v_cvt_pk_bf16_f32 %0, %1, %2" : "=v"(r) : "v"(a), "v"(b));
  return r;
}
#endif

__device__ __forceinline__ u16 f2bf(float f) { return (u16)pack_bf16(f, 0.f); }

// ---------------------------------------------------------------------------
// TDM: 1D tile copy global -> LDS via tensor_load_to_lds (D# per ISA 08).
//   group0: count=1 | lds_addr | global_addr[56:0] | type=2
//   group1: data_size=1(2B), tensor_dim0=tile_dim0=nelem, tensor/tile_dim1=1
// Issued once per block by wave 0; completion via s_wait_tensorcnt.
// ---------------------------------------------------------------------------
#if __has_builtin(__builtin_amdgcn_tensor_load_to_lds)
#define HAVE_TDM 1
__device__ __forceinline__ void tdm_copy_to_lds(const void* gsrc, u32 lds_off,
                                                u32 nelem_u16) {
  unsigned long long ga = (unsigned long long)(uintptr_t)gsrc;
  u32x4 g0;
  g0[0] = 1u;                                           // count=1 (valid D#)
  g0[1] = lds_off;                                      // lds_addr
  g0[2] = (u32)ga;                                      // global_addr[31:0]
  g0[3] = (u32)((ga >> 32) & 0x01FFFFFFu) | 0x80000000u; // addr[56:32] | type=2
  i32x8 g1;
  g1[0] = (int)(1u << 16);                              // data_size=1 (2 bytes)
  g1[1] = (int)((nelem_u16 & 0xFFFFu) << 16);           // tensor_dim0[15:0]
  g1[2] = (int)(((nelem_u16 >> 16) & 0xFFFFu) | (1u << 16)); // dim0 hi | tensor_dim1=1
  g1[3] = (int)((nelem_u16 & 0xFFFFu) << 16);           // tile_dim0 = nelem
  g1[4] = 1;                                            // tile_dim1=1, tile_dim2=0
  g1[5] = (int)nelem_u16;                               // tensor_dim0_stride lo
  g1[6] = 0;
  g1[7] = 0;
  i32x4 z4 = {0, 0, 0, 0};
#if __clang_major__ >= 23
  i32x8 z8 = {0, 0, 0, 0, 0, 0, 0, 0};
  __builtin_amdgcn_tensor_load_to_lds(g0, g1, z4, z4, z8, 0);
#else
  __builtin_amdgcn_tensor_load_to_lds(g0, g1, z4, z4, 0);
#endif
}
#endif

// A fragment (16-bit, 16x32): lane L holds row M=L&15.
// lanes 0-15: K = ko..ko+7 and ko+16..ko+23 ; lanes 16-31: +8 shift.
__device__ __forceinline__ bf16x16 ldfragA(const u16* base, int stride, int lane, int ko) {
  int M  = lane & 15;
  int hk = (lane & 16) ? 8 : 0;
  const u16* p = base + (long)M * stride + ko + hk;
  ABu f;
  f.q[0] = *(const uint4*)(p);
  f.q[1] = *(const uint4*)(p + 16);
  return f.v;
}

// B fragment (16-bit, 32x16): lane holds column N=lane&15, same K order as A
// (lanes 0-15: K=ko..ko+7, ko+16..ko+23 ; lanes 16-31: +8 shift). Row-major w[d][c]
// (d=N, c=K) makes these contiguous 16-byte runs.
__device__ __forceinline__ bf16x16 ldfragB(const u16* base, int stride, int lane, int ko) {
  int N  = lane & 15;
  int hk = (lane & 16) ? 8 : 0;
  const u16* p = base + (long)N * stride + ko + hk;
  ABu f;
  f.q[0] = *(const uint4*)(p);
  f.q[1] = *(const uint4*)(p + 16);
  return f.v;
}

__device__ __forceinline__ f32x8 wmma_bf16(bf16x16 a, bf16x16 b, f32x8 c) {
  return __builtin_amdgcn_wmma_f32_16x16x32_bf16(false, a, false, b, (short)0, c, false, false);
}

// ---------------------------------------------------------------------------
// Kernel 1: convert weights to bf16, zero accumulators.
// ---------------------------------------------------------------------------
__global__ __launch_bounds__(256) void prep_kernel(
    const float* __restrict__ w1, const float* __restrict__ w2,
    const float* __restrict__ wc,
    u16* __restrict__ w1bf, u16* __restrict__ w2bf, u16* __restrict__ wcbf,
    float* __restrict__ sc, float* __restrict__ sr) {
  long i = (long)blockIdx.x * 256 + threadIdx.x;
  long stride = (long)gridDim.x * 256;
  for (long k = i; k < CZ * CZ; k += stride) {
    w1bf[k] = f2bf(w1[k]);
    w2bf[k] = f2bf(w2[k]);
  }
  for (long k = i; k < (long)CS * KCAT; k += stride) wcbf[k] = f2bf(wc[k]);
  for (long k = i; k < (long)L_DIM * CZ; k += stride) { sc[k] = 0.f; sr[k] = 0.f; }
}

// ---------------------------------------------------------------------------
// Kernel 2: mask norms. blocks 0..1023 -> row sums (norm_c), 1024..2047 -> col.
// ---------------------------------------------------------------------------
__global__ __launch_bounds__(256) void norm_kernel(
    const float* __restrict__ mask, float* __restrict__ normc,
    float* __restrict__ normr) {
  __shared__ float red[256];
  int b = blockIdx.x, t = threadIdx.x;
  float s = 0.f;
  if (b < L_DIM) {
    const float* row = mask + (long)b * L_DIM;
    for (int i = t; i < L_DIM; i += 256) s += row[i];
  } else {
    int m = b - L_DIM;
    for (int i = t; i < L_DIM; i += 256) s += mask[(long)i * L_DIM + m];
  }
  red[t] = s;
  __syncthreads();
  for (int o = 128; o > 0; o >>= 1) {
    if (t < o) red[t] += red[t + o];
    __syncthreads();
  }
  if (t == 0) {
    float v = red[0];
    v = v < 1.f ? 1.f : v;
    if (b < L_DIM) normc[b] = v; else normr[b - L_DIM] = v;
  }
}

// ---------------------------------------------------------------------------
// Kernel 3: fused LN -> Linear -> ReLU -> Linear -> mask -> row/col sums.
// grid = (8 m-groups of 128, 64 l-groups of 16), 256 threads = 8 waves.
// Wave w owns m-strip [blockIdx.x*128 + 16w, +16) and loops 16 values of l.
// Weights (w1|w2, contiguous 64KB) staged to LDS with one TDM descriptor.
// ---------------------------------------------------------------------------
__global__ __launch_bounds__(256) void pair_kernel(
    const float* __restrict__ s_z, const float* __restrict__ mask,
    const float* __restrict__ ln_g, const float* __restrict__ ln_b,
    const float* __restrict__ b1, const float* __restrict__ b2,
    const u16* __restrict__ w1bf, const u16* __restrict__ w2bf,
    float* __restrict__ sc, float* __restrict__ sr) {
  extern __shared__ u16 smem_u[];
  u16*   w1s   = smem_u;                       // 128*128 bf16 = 32 KB (offset 0)
  u16*   w2s   = w1s + CZ * CZ;                // 32 KB
  u16*   ztAll = w2s + CZ * CZ;                // 8 waves * 16*128 bf16 = 32 KB
  u16*   htAll = ztAll + 8 * 16 * CZ;          // 32 KB
  float* scS   = (float*)(htAll + 8 * 16 * CZ); // 16*128 f32 = 8 KB

  int t = threadIdx.x;
  int wv = t >> 5, lane = t & 31;

#ifdef HAVE_TDM
  // One DMA: w1bf..w2bf are contiguous (2*128*128 u16 = 64 KB) -> LDS offset 0.
  if (wv == 0) {
    tdm_copy_to_lds(w1bf, 0u, 2u * CZ * CZ);
    __builtin_amdgcn_s_wait_tensorcnt(0);
  }
#else
  for (int i = t; i < (CZ * CZ) / 8; i += 256) {
    ((uint4*)w1s)[i] = ((const uint4*)w1bf)[i];
    ((uint4*)w2s)[i] = ((const uint4*)w2bf)[i];
  }
#endif
  for (int i = t; i < 16 * CZ; i += 256) scS[i] = 0.f;
  __syncthreads();

  u16* zt = ztAll + wv * 16 * CZ;   // wave-private LN'd tile (16 pairs x 128)
  u16* ht = htAll + wv * 16 * CZ;   // wave-private hidden tile
  int m0    = blockIdx.x * 128 + wv * 16;
  int l0    = blockIdx.y * 16;
  int colL  = lane & 15;
  int half8 = (lane & 16) ? 8 : 0;  // C/D fragment row offset for upper lanes

  float4 gv = *(const float4*)(ln_g + lane * 4);
  float4 bv = *(const float4*)(ln_b + lane * 4);
  float b1v[8], b2v[8];
#pragma unroll
  for (int nt = 0; nt < 8; ++nt) {
    b1v[nt] = b1[nt * 16 + colL];
    b2v[nt] = b2[nt * 16 + colL];
  }

  f32x8 accR[8] = {};  // running row-sum (s_s_r) fragments across the l loop

  for (int il = 0; il < 16; ++il) {
    int l = l0 + il;
    long base = ((long)l * L_DIM + m0) * CZ;

    // ---- load 16 pair-rows (lane*4 channels each) and LayerNorm ----
    float4 xr[16];
#pragma unroll
    for (int j = 0; j < 16; ++j)
      xr[j] = *(const float4*)(s_z + base + j * CZ + lane * 4);
    if (il < 15) {
#pragma unroll
      for (int j = 0; j < 16; ++j)
        __builtin_prefetch(s_z + base + (long)L_DIM * CZ + j * CZ + lane * 4, 0, 1);
    }
#pragma unroll
    for (int j = 0; j < 16; ++j) {
      float4 x = xr[j];
      float s  = x.x + x.y + x.z + x.w;
      float ss = x.x * x.x + x.y * x.y + x.z * x.z + x.w * x.w;
#pragma unroll
      for (int off = 16; off >= 1; off >>= 1) {
        s  += __shfl_xor(s, off, 32);
        ss += __shfl_xor(ss, off, 32);
      }
      float mu   = s * (1.f / 128.f);
      float rstd = rsqrtf(ss * (1.f / 128.f) - mu * mu + 1e-5f);
      float n0 = (x.x - mu) * rstd * gv.x + bv.x;
      float n1 = (x.y - mu) * rstd * gv.y + bv.y;
      float n2 = (x.z - mu) * rstd * gv.z + bv.z;
      float n3 = (x.w - mu) * rstd * gv.w + bv.w;
      uint2 pk;
      pk.x = pack_bf16(n0, n1);
      pk.y = pack_bf16(n2, n3);
      *(uint2*)(zt + j * CZ + lane * 4) = pk;
    }
    WAVE_BARRIER();

    // ---- layer 1: (16x128) x (128x128), bf16 WMMA, fp32 accum ----
    f32x8 c1[8] = {};
#pragma unroll
    for (int ko = 0; ko < CZ; ko += 32) {
      bf16x16 a = ldfragA(zt, CZ, lane, ko);
#pragma unroll
      for (int nt = 0; nt < 8; ++nt) {
        bf16x16 b = ldfragB(w1s + nt * 16 * CZ, CZ, lane, ko);
        c1[nt] = wmma_bf16(a, b, c1[nt]);
      }
    }
    // bias + ReLU -> bf16 hidden tile (row-major scatter, b16 stores)
#pragma unroll
    for (int nt = 0; nt < 8; ++nt) {
      int cc = nt * 16 + colL;
#pragma unroll
      for (int r = 0; r < 8; ++r) {
        float h = c1[nt][r] + b1v[nt];
        h = h > 0.f ? h : 0.f;
        ht[(r + half8) * CZ + cc] = f2bf(h);
      }
    }
    WAVE_BARRIER();

    // ---- layer 2 ----
    f32x8 c2[8] = {};
#pragma unroll
    for (int ko = 0; ko < CZ; ko += 32) {
      bf16x16 a = ldfragA(ht, CZ, lane, ko);
#pragma unroll
      for (int nt = 0; nt < 8; ++nt) {
        bf16x16 b = ldfragB(w2s + nt * 16 * CZ, CZ, lane, ko);
        c2[nt] = wmma_bf16(a, b, c2[nt]);
      }
    }

    // ---- epilogue: bias, mask, running reductions ----
    const float* mrow = mask + (long)l * L_DIM + m0;
    float mv[8];
#pragma unroll
    for (int r = 0; r < 8; ++r) mv[r] = mrow[r + half8];
#pragma unroll
    for (int nt = 0; nt < 8; ++nt) {
      float csum = 0.f;
#pragma unroll
      for (int r = 0; r < 8; ++r) {
        float o = (c2[nt][r] + b2v[nt]) * mv[r];
        accR[nt][r] += o;   // s_s_r partial: same m rows every iteration
        csum += o;          // s_s_c partial: sum over this tile's 8 rows
      }
      csum += __shfl_xor(csum, 16, 32);  // combine both row-halves
      if (lane < 16) atomicAdd(&scS[il * CZ + nt * 16 + lane], csum);
    }
  }

  // flush s_s_r register partials (sum over this block's 16 l's)
#pragma unroll
  for (int nt = 0; nt < 8; ++nt) {
    int cc = nt * 16 + colL;
#pragma unroll
    for (int r = 0; r < 8; ++r)
      atomicAdd(&sr[(long)(m0 + r + half8) * CZ + cc], accR[nt][r]);
  }

  __syncthreads();
  // flush s_s_c LDS partials (sum over this block's 128 m's)
  for (int i = t; i < 16 * CZ; i += 256)
    atomicAdd(&sc[(long)(l0 + (i >> 7)) * CZ + (i & 127)], scS[i]);
}

// ---------------------------------------------------------------------------
// Kernel 4: build bf16 concatenated features [sc/nc | sr/nr | s_s_in].
// ---------------------------------------------------------------------------
__global__ __launch_bounds__(256) void cat_kernel(
    const float* __restrict__ sc, const float* __restrict__ sr,
    const float* __restrict__ normc, const float* __restrict__ normr,
    const float* __restrict__ s_s_in, u16* __restrict__ catbf) {
  long i = (long)blockIdx.x * 256 + threadIdx.x;
  if (i >= (long)L_DIM * KCAT) return;
  int l = (int)(i / KCAT), c = (int)(i % KCAT);
  float v;
  if (c < CZ)            v = sc[l * CZ + c] / normc[l];
  else if (c < 2 * CZ)   v = sr[l * CZ + (c - CZ)] / normr[l];
  else                   v = s_s_in[(long)l * CS + (c - 2 * CZ)];
  catbf[i] = f2bf(v);
}

// ---------------------------------------------------------------------------
// Kernel 5: out[l,d] = cat[l,:] . wc[d,:] + bc[d]  (M=N=1024, K=1280, WMMA).
// One 16x16 tile per wave, 40 K-steps, fully unrolled.
// ---------------------------------------------------------------------------
__global__ __launch_bounds__(256) void out_kernel(
    const u16* __restrict__ catbf, const u16* __restrict__ wcbf,
    const float* __restrict__ bc, float* __restrict__ out) {
  int t = threadIdx.x, wv = t >> 5, lane = t & 31;
  int tile = blockIdx.x * 8 + wv;       // 64x64 tiles of 16x16
  int mT = tile >> 6, nT = tile & 63;
  int M0 = mT * 16, N0 = nT * 16;
  int colL = lane & 15, half8 = (lane & 16) ? 8 : 0;
  const u16* abase = catbf + (long)M0 * KCAT;
  const u16* bbase = wcbf + (long)N0 * KCAT;
  f32x8 acc = {};
#pragma unroll
  for (int ko = 0; ko < KCAT; ko += 32) {
    bf16x16 a = ldfragA(abase, KCAT, lane, ko);
    bf16x16 b = ldfragB(bbase, KCAT, lane, ko);
    acc = wmma_bf16(a, b, acc);
  }
  float bcv = bc[N0 + colL];
#pragma unroll
  for (int r = 0; r < 8; ++r)
    out[(long)(M0 + r + half8) * CS + N0 + colL] = acc[r] + bcv;
}

// ---------------------------------------------------------------------------
extern "C" void kernel_launch(void* const* d_in, const int* in_sizes, int n_in,
                              void* d_out, int out_size, void* d_ws, size_t ws_size,
                              hipStream_t stream) {
  const float* s_z    = (const float*)d_in[0];
  const float* s_s_in = (const float*)d_in[1];
  const float* mask   = (const float*)d_in[2];
  const float* ln_g   = (const float*)d_in[3];
  const float* ln_b   = (const float*)d_in[4];
  const float* w1     = (const float*)d_in[5];
  const float* b1     = (const float*)d_in[6];
  const float* w2     = (const float*)d_in[7];
  const float* b2     = (const float*)d_in[8];
  const float* wc     = (const float*)d_in[9];
  const float* bc     = (const float*)d_in[10];
  float* out = (float*)d_out;

  char* ws = (char*)d_ws;
  size_t off = 0;
  auto carve = [&](size_t bytes) { void* p = ws + off; off += (bytes + 255) & ~size_t(255); return p; };
  float* sc    = (float*)carve(sizeof(float) * L_DIM * CZ);   // 512 KB
  float* sr    = (float*)carve(sizeof(float) * L_DIM * CZ);   // 512 KB
  float* normc = (float*)carve(sizeof(float) * L_DIM);
  float* normr = (float*)carve(sizeof(float) * L_DIM);
  u16* w1bf    = (u16*)carve(sizeof(u16) * CZ * CZ);          // contiguous with w2bf:
  u16* w2bf    = (u16*)carve(sizeof(u16) * CZ * CZ);          //  (32KB each, 256B aligned)
  u16* wcbf    = (u16*)carve(sizeof(u16) * CS * KCAT);        // 2.62 MB
  u16* catbf   = (u16*)carve(sizeof(u16) * L_DIM * KCAT);     // 2.62 MB
  (void)ws_size; (void)in_sizes; (void)n_in; (void)out_size;

  // 1) weight conversion + accumulator zeroing (covers 1,310,720 elems in 1 pass)
  prep_kernel<<<5120, 256, 0, stream>>>(w1, w2, wc, w1bf, w2bf, wcbf, sc, sr);

  // 2) mask norms
  norm_kernel<<<2048, 256, 0, stream>>>(mask, normc, normr);

  // 3) fused pair pipeline
  size_t smem = (size_t)(2 * CZ * CZ + 2 * 8 * 16 * CZ) * sizeof(u16) +
                (size_t)(16 * CZ) * sizeof(float);  // 139,264 B
  pair_kernel<<<dim3(8, 64), 256, smem, stream>>>(
      s_z, mask, ln_g, ln_b, b1, b2, w1bf, w2bf, sc, sr);

  // 4) concat features
  cat_kernel<<<(L_DIM * KCAT + 255) / 256, 256, 0, stream>>>(
      sc, sr, normc, normr, s_s_in, catbf);

  // 5) final projection
  out_kernel<<<(64 * 64) / 8, 256, 0, stream>>>(catbf, wcbf, bc, out);
}
